// Encoder_86234353369864
// MI455X (gfx1250) — matrix operand order repaired and checked by reading, not compiled, not run on hardware
//
#include <hip/hip_runtime.h>

typedef __attribute__((ext_vector_type(16))) __bf16        v16bf;
typedef __attribute__((ext_vector_type(8)))  float         v8f;
typedef __attribute__((ext_vector_type(8)))  unsigned int  v8u;

#define T_SEQ   2048
#define B_BATCH 64
#define D_IN    64
#define H_DIM   128

__device__ __forceinline__ unsigned short f2bf(float x) {
    unsigned int u = __float_as_uint(x);
    unsigned int r = (u + 0x7FFFu + ((u >> 16) & 1u)) >> 16;  // RNE
    return (unsigned short)r;
}
__device__ __forceinline__ unsigned int pack2(float a, float b) {
    return (unsigned int)f2bf(a) | ((unsigned int)f2bf(b) << 16);
}
__device__ __forceinline__ float fast_sigmoid(float x) {
    return __builtin_amdgcn_rcpf(1.0f + __expf(-x));
}
__device__ __forceinline__ float fast_tanh(float x) {
    return 1.0f - 2.0f * __builtin_amdgcn_rcpf(1.0f + __expf(2.0f * x));
}

// One workgroup = 16 batch rows, 8 wave32s. Wave w owns h-columns
// [16w, 16w+16) of all four gates (i,f,g,o). Weights are register-resident
// bf16 WMMA B-fragments. h and x_t tiles are staged in ping-pong LDS
// buffers (one barrier per step); the x_{t+1} global fetch is software-
// pipelined one full iteration ahead so its latency is hidden.
__global__ void __launch_bounds__(256)
lstm_encoder_kernel(const float* __restrict__ x,    // (B, T, D)
                    const float* __restrict__ Wih,  // (4H, D)
                    const float* __restrict__ Whh,  // (4H, H)
                    const float* __restrict__ bih,  // (4H,)
                    const float* __restrict__ bhh,  // (4H,)
                    const float* __restrict__ h0,   // (B, H)
                    const float* __restrict__ c0,   // (B, H)
                    float* __restrict__ out)        // (B, T, H)
{
    __shared__ alignas(16) unsigned short hbuf[2][16 * H_DIM];  // 2 x 4 KB
    __shared__ alignas(16) unsigned short xbuf[2][16 * D_IN];   // 2 x 2 KB

    const int tid  = threadIdx.x;
    const int wave = tid >> 5;
    const int lane = tid & 31;
    const int half = lane >> 4;   // K-half selector for A/B frags
    const int ln   = lane & 15;   // N (column) within tile
    const int b0   = blockIdx.x * 16;

    // ---- Register-resident B fragments (weights, bf16) ----
    // B[k][n] = W[n][k]; lane ln holds column n, VGPR j holds K = half*16 + 2j(+1).
    v8u bh[4][4];  // [gate][kchunk of 32] for W_hh^T (K = 128)
    v8u bx[4][2];  // [gate][kchunk of 32] for W_ih^T (K = 64)
#pragma unroll
    for (int g = 0; g < 4; ++g) {
        const int col = g * H_DIM + wave * 16 + ln;
        const float* wr = Whh + (size_t)col * H_DIM;
#pragma unroll
        for (int kc = 0; kc < 4; ++kc) {
            const float* p = wr + kc * 32 + half * 16;
#pragma unroll
            for (int i = 0; i < 8; ++i)
                bh[g][kc][i] = pack2(p[2 * i], p[2 * i + 1]);
        }
        const float* wr2 = Wih + (size_t)col * D_IN;
#pragma unroll
        for (int kc = 0; kc < 2; ++kc) {
            const float* p = wr2 + kc * 32 + half * 16;
#pragma unroll
            for (int i = 0; i < 8; ++i)
                bx[g][kc][i] = pack2(p[2 * i], p[2 * i + 1]);
        }
    }

    // ---- Bias (depends only on gate column) ----
    float bias[4];
#pragma unroll
    for (int g = 0; g < 4; ++g) {
        const int idx = g * H_DIM + wave * 16 + ln;
        bias[g] = bih[idx] + bhh[idx];
    }

    // ---- Cell state tile (fp32): element r <-> (row M=r+half*8, col hcol) ----
    const int hcol = wave * 16 + ln;
    float cst[8];
#pragma unroll
    for (int r = 0; r < 8; ++r)
        cst[r] = c0[(size_t)(b0 + r + half * 8) * H_DIM + hcol];

    // ---- Per-thread x staging assignment (16 rows x 64 cols, float4 each) ----
    const int xr = tid >> 4, xd = (tid & 15) * 4;
    const float* xrow = x + (size_t)(b0 + xr) * T_SEQ * D_IN + xd;

    // ---- Seed: hbuf[0] <- h0, xbuf[0] <- x_0, xreg <- x_1 (in flight) ----
    for (int e = tid; e < 16 * H_DIM; e += 256)
        hbuf[0][e] = f2bf(h0[(size_t)(b0 + (e >> 7)) * H_DIM + (e & 127)]);
    {
        const float4 v = *(const float4*)(xrow + 0 * D_IN);
        unsigned int* q = (unsigned int*)xbuf[0] + (xr * D_IN + xd) / 2;
        q[0] = pack2(v.x, v.y);
        q[1] = pack2(v.z, v.w);
    }
    float4 xreg = *(const float4*)(xrow + (size_t)1 * D_IN);  // x_{t+1} pipeline reg
    __syncthreads();

    for (int t = 0; t < T_SEQ; ++t) {
        const int p = t & 1, q = p ^ 1;

        // ---- A fragments from LDS ping buffer (16-bit A layout) ----
        v8u ah[4], ax[2];
        {
            const uint4* ph = (const uint4*)(hbuf[p] + ln * H_DIM);
#pragma unroll
            for (int kc = 0; kc < 4; ++kc) {
                uint4 lo = ph[(kc * 32 + half * 16) >> 3];
                uint4 hi = ph[(kc * 32 + half * 16 + 8) >> 3];
                ah[kc][0] = lo.x; ah[kc][1] = lo.y; ah[kc][2] = lo.z; ah[kc][3] = lo.w;
                ah[kc][4] = hi.x; ah[kc][5] = hi.y; ah[kc][6] = hi.z; ah[kc][7] = hi.w;
            }
            const uint4* px = (const uint4*)(xbuf[p] + ln * D_IN);
#pragma unroll
            for (int kc = 0; kc < 2; ++kc) {
                uint4 lo = px[(kc * 32 + half * 16) >> 3];
                uint4 hi = px[(kc * 32 + half * 16 + 8) >> 3];
                ax[kc][0] = lo.x; ax[kc][1] = lo.y; ax[kc][2] = lo.z; ax[kc][3] = lo.w;
                ax[kc][4] = hi.x; ax[kc][5] = hi.y; ax[kc][6] = hi.z; ax[kc][7] = hi.w;
            }
        }

        // ---- Drain pipelined x_{t+1} into pong buffer; issue x_{t+2} ----
        {
            unsigned int* qx = (unsigned int*)xbuf[q] + (xr * D_IN + xd) / 2;
            qx[0] = pack2(xreg.x, xreg.y);
            qx[1] = pack2(xreg.z, xreg.w);
            const int tnn = (t + 2 < T_SEQ) ? (t + 2) : (T_SEQ - 1);
            xreg = *(const float4*)(xrow + (size_t)tnn * D_IN);
        }

        // ---- gates = x_t @ Wih^T + h @ Whh^T + bias (24 WMMAs / wave) ----
        v8f acc[4];
#pragma unroll
        for (int g = 0; g < 4; ++g) {
            v8f a;
#pragma unroll
            for (int r = 0; r < 8; ++r) a[r] = bias[g];
#pragma unroll
            for (int kc = 0; kc < 2; ++kc)
                a = __builtin_amdgcn_wmma_f32_16x16x32_bf16(
                        false, __builtin_bit_cast(v16bf, ax[kc]),
                        false, __builtin_bit_cast(v16bf, bx[g][kc]),
                        (short)0, a, false, false);
#pragma unroll
            for (int kc = 0; kc < 4; ++kc)
                a = __builtin_amdgcn_wmma_f32_16x16x32_bf16(
                        false, __builtin_bit_cast(v16bf, ah[kc]),
                        false, __builtin_bit_cast(v16bf, bh[g][kc]),
                        (short)0, a, false, false);
            acc[g] = a;
        }

        // ---- Elementwise LSTM cell (fp32) ----
        unsigned short hnew[8];
#pragma unroll
        for (int r = 0; r < 8; ++r) {
            const float iv = fast_sigmoid(acc[0][r]);
            const float fv = fast_sigmoid(acc[1][r]);
            const float gv = fast_tanh(acc[2][r]);
            const float ov = fast_sigmoid(acc[3][r]);
            const float cn = fv * cst[r] + iv * gv;
            cst[r] = cn;
            const float hv = ov * fast_tanh(cn);
            hnew[r] = f2bf(hv);
            out[((size_t)(b0 + r + half * 8) * T_SEQ + t) * H_DIM + hcol] = hv;
        }
#pragma unroll
        for (int r = 0; r < 8; ++r)
            hbuf[q][(r + half * 8) * H_DIM + hcol] = hnew[r];

        __syncthreads();  // pong buffers (h_{t+1}, x_{t+1}) visible for next step
    }
}

extern "C" void kernel_launch(void* const* d_in, const int* in_sizes, int n_in,
                              void* d_out, int out_size, void* d_ws, size_t ws_size,
                              hipStream_t stream) {
    const float* x   = (const float*)d_in[0];
    const float* Wih = (const float*)d_in[1];
    const float* Whh = (const float*)d_in[2];
    const float* bih = (const float*)d_in[3];
    const float* bhh = (const float*)d_in[4];
    const float* h0  = (const float*)d_in[5];
    const float* c0  = (const float*)d_in[6];
    float* out = (float*)d_out;

    dim3 grid(B_BATCH / 16);  // 4 persistent workgroups (one M-tile each)
    dim3 block(256);          // 8 wave32s
    hipLaunchKernelGGL(lstm_encoder_kernel, grid, block, 0, stream,
                       x, Wih, Whh, bih, bhh, h0, c0, out);
}
